// WiKG_81862076662087
// MI455X (gfx1250) — matrix operand order repaired and checked by reading, not compiled, not run on hardware
//
#include <hip/hip_runtime.h>
#include <hip/hip_bf16.h>

typedef __attribute__((ext_vector_type(16))) _Float16 v16h;
typedef __attribute__((ext_vector_type(8)))  float    v8f;
typedef __attribute__((ext_vector_type(4)))  float    f32x4;
typedef __attribute__((ext_vector_type(4)))  unsigned int u32x4;
typedef __attribute__((ext_vector_type(4)))  int      i32x4;
typedef __attribute__((ext_vector_type(8)))  int      i32x8;

#define N_NODES 8192
#define DIN 384
#define DH 512
#define TOPK 6
#define ATT_SCALE 0.04419417382415922f  /* 512^-0.5 */
#define CHUNK 512
#define NEG_BIG (-1.0e30f)

// ---------------------------------------------------------------------------
// WMMA fragment loaders (CDNA5 16-bit layouts, CDNA5 ISA §7.12.2)
// A 16x32 f16: lane L: row=L%16, half=L/16;
//   halfs[0..7]  = A[row][k0 + half*8 .. +7]
//   halfs[8..15] = A[row][k0 + 16 + half*8 .. +7]
// B 32x16 f16: lane L: col=L%16, half=L/16;
//   halfs[0..7] = B[k0 + half*8 .. +7][col], halfs[8..15] = B[k0 + 16 + half*8 .. +7][col]   (we feed Bt rows)
// ---------------------------------------------------------------------------
__device__ __forceinline__ v16h load_frag_a(const _Float16* rowPtr, int k0, int half) {
  union { f32x4 f[2]; v16h h; } u;
  u.f[0] = *(const f32x4*)(rowPtr + k0 + half * 8);
  u.f[1] = *(const f32x4*)(rowPtr + k0 + 16 + half * 8);
  return u.h;
}
__device__ __forceinline__ v16h load_frag_b(const _Float16* rowPtr, int k0, int half) {
  union { f32x4 f[2]; v16h h; } u;
  u.f[0] = *(const f32x4*)(rowPtr + k0 + half * 8);
  u.f[1] = *(const f32x4*)(rowPtr + k0 + 16 + half * 8);
  return u.h;
}

// ---------------------------------------------------------------------------
// TDM staging: load a 16-row x 512-half f16 tile into LDS with 4-DWORD padding
// inserted after every 256 DWORDs (one row) => 520-half LDS row stride.
// D# encoding per CDNA5 ISA (group0: count=1, type=2;
// group1: data_size=2B, pad_enable, pad_interval=7 (256 dw), pad_amount=3 (4 dw),
// tile 512x16, tensor_dim0_stride=512).
// ---------------------------------------------------------------------------
__device__ __forceinline__ void tdm_stage_rows(const _Float16* gsrc, void* ldsDst) {
  const unsigned long long ga = (unsigned long long)(uintptr_t)gsrc;
  const unsigned int lds = (unsigned int)(uintptr_t)ldsDst;  // addr[31:0] = LDS offset
  u32x4 g0;
  g0[0] = 1u;                                                // count=1, user mode
  g0[1] = lds;                                               // lds_addr
  g0[2] = (unsigned int)(ga & 0xFFFFFFFFu);                  // global_addr[31:0]
  g0[3] = (unsigned int)((ga >> 32) & 0x01FFFFFFu) | (2u << 30);  // [56:32] | type=2
  i32x8 g1;
  g1[0] = (int)((1u << 16) | (1u << 20) | (7u << 22) | (3u << 25));
  g1[1] = (int)(512u << 16);   // tensor_dim0[15:0] in bits[31:16]
  g1[2] = (int)(8192u << 16);  // tensor_dim0[31:16]=0 | tensor_dim1[15:0]
  g1[3] = (int)(512u << 16);   // tensor_dim1[31:16]=0 | tile_dim0=512
  g1[4] = 16;                  // tile_dim1=16 | tile_dim2=0
  g1[5] = 512;                 // tensor_dim0_stride[31:0]
  g1[6] = 0;                   // stride hi | tensor_dim1_stride lo
  g1[7] = 0;
  const i32x4 z4 = {0, 0, 0, 0};
#if __has_include(<hip/amd_detail/amd_gfx1250_TDM.h>)
  const i32x8 z8 = {0, 0, 0, 0, 0, 0, 0, 0};
  __builtin_amdgcn_tensor_load_to_lds(g0, g1, z4, z4, z8, 0);
#else
  __builtin_amdgcn_tensor_load_to_lds(g0, g1, z4, z4, 0);
#endif
}

// ---------------------------------------------------------------------------
// Generic WMMA GEMM: C[M,N] = act(A[M,K] @ Bt[N,K]^T + bias), f16 in, f32 out.
// Wave computes 16x64 (4 accumulators). K-loop is a 2x-unrolled ping-pong:
// fragments for step k+32 load into the alternate buffer while WMMAs consume
// the current one -> partial loadcnt waits, zero register-rotation copies.
// Requires K % 64 == 0 (384, 512 here). blockDim=256 (8 waves),
// grid=(N/64, M/128). act: 0=none, 1=leaky_relu(0.01). acc: C += result.
// ---------------------------------------------------------------------------
__global__ __launch_bounds__(256) void gemm_wmma(
    const _Float16* __restrict__ A, const _Float16* __restrict__ Bt,
    const float* __restrict__ bias, float* __restrict__ C,
    _Float16* __restrict__ C16, float c16scale,
    int M, int N, int K, int act, int acc) {
  const int wave = threadIdx.x >> 5;
  const int lane = threadIdx.x & 31;
  const int half = lane >> 4;
  const int l16  = lane & 15;
  const int mtile = blockIdx.y * 8 + wave;
  const int n0 = blockIdx.x * 64;
  const _Float16* aRow = A + (size_t)(mtile * 16 + l16) * K;
  const _Float16* bRow[4];
#pragma unroll
  for (int nt = 0; nt < 4; nt++) bRow[nt] = Bt + (size_t)(n0 + nt * 16 + l16) * K;

  v8f cacc[4] = {v8f{0.f}, v8f{0.f}, v8f{0.f}, v8f{0.f}};
  v16h aF[2];
  v16h bF[2][4];
  aF[0] = load_frag_a(aRow, 0, half);
#pragma unroll
  for (int nt = 0; nt < 4; nt++) bF[0][nt] = load_frag_b(bRow[nt], 0, half);

  for (int k0 = 0; k0 < K; k0 += 64) {
    const int k1 = k0 + 32;                      // K % 64 == 0 -> always valid
    aF[1] = load_frag_a(aRow, k1, half);
#pragma unroll
    for (int nt = 0; nt < 4; nt++) bF[1][nt] = load_frag_b(bRow[nt], k1, half);
#pragma unroll
    for (int nt = 0; nt < 4; nt++)
      cacc[nt] = __builtin_amdgcn_wmma_f32_16x16x32_f16(
          false, aF[0], false, bF[0][nt], (short)0, cacc[nt], false, false);
    const int k2 = (k0 + 64 < K) ? k0 + 64 : 0;  // clamped dead prefetch on last
    aF[0] = load_frag_a(aRow, k2, half);
#pragma unroll
    for (int nt = 0; nt < 4; nt++) bF[0][nt] = load_frag_b(bRow[nt], k2, half);
#pragma unroll
    for (int nt = 0; nt < 4; nt++)
      cacc[nt] = __builtin_amdgcn_wmma_f32_16x16x32_f16(
          false, aF[1], false, bF[1][nt], (short)0, cacc[nt], false, false);
  }
#pragma unroll
  for (int nt = 0; nt < 4; nt++) {
    const int col = n0 + nt * 16 + l16;
    const float bv = bias ? bias[col] : 0.f;
#pragma unroll
    for (int v = 0; v < 8; v++) {
      const int row = mtile * 16 + v + 8 * half;
      float val = cacc[nt][v] + bv;
      if (act == 1) val = (val > 0.f) ? val : 0.01f * val;
      const size_t o = (size_t)row * N + col;
      if (acc) { if (C) C[o] += val; }
      else     { if (C) C[o]  = val; }
      if (C16) C16[o] = (_Float16)(val * c16scale);
    }
  }
}

// ---------------------------------------------------------------------------
// small helpers
// ---------------------------------------------------------------------------
__global__ void zero_f32(float* p, int n) {
  int i = blockIdx.x * blockDim.x + threadIdx.x;
  if (i < n) p[i] = 0.f;
}
__global__ void to_f16(const float* __restrict__ X, _Float16* __restrict__ Y, int n) {
  int i = blockIdx.x * blockDim.x + threadIdx.x;
  if (i < n) Y[i] = (_Float16)X[i];
}
// W[K,N] f32 -> Wt[N,K] f16
__global__ void transpose_f16(const float* __restrict__ W, _Float16* __restrict__ Wt,
                              int K, int N) {
  int idx = blockIdx.x * blockDim.x + threadIdx.x;
  if (idx >= K * N) return;
  int n = idx / K, k = idx % K;
  Wt[idx] = (_Float16)W[(size_t)k * N + n];
}
// partial column sums, rows in chunks of 64; blockDim=512 (one thread per col)
__global__ __launch_bounds__(512) void colsum_kernel(const float* __restrict__ X,
                                                     float* __restrict__ colsum) {
  const int col = threadIdx.x;
  const int r0 = blockIdx.x * 64;
  float s = 0.f;
#pragma unroll 4
  for (int i = 0; i < 64; i++) s += X[(size_t)(r0 + i) * DH + col];
  atomicAdd(&colsum[col], s);
}
// x = (x + colmean)*0.5 in place, also emit f16 copy. grid=N_NODES, block=DH
__global__ __launch_bounds__(512) void apply_mean(float* __restrict__ X,
                                                  _Float16* __restrict__ X16,
                                                  const float* __restrict__ colsum) {
  const size_t o = (size_t)blockIdx.x * DH + threadIdx.x;
  float v = (X[o] + colsum[threadIdx.x] * (1.f / (float)N_NODES)) * 0.5f;
  X[o] = v;
  X16[o] = (_Float16)v;
}

// ---------------------------------------------------------------------------
// Fused attention + top-k + neighbor aggregation.
// One block = 16 rows, 16 waves (blockDim=512).
//   16 A rows (e_h*SCALE, f16) staged into padded LDS via the Tensor Data
//   Mover; B fragments stream e_t f16 rows straight from L2 (A*B^T => both
//   row-major, two b128 loads per fragment). Each wave computes two adjacent
//   16x16 logit tiles (shared A frag, 2 accumulators); K-loop is a 2x-unrolled
//   ping-pong (no rotation copies). Logits chunked (16x512 f32) through LDS;
//   per-row running top-6 via register insertion + wave32 butterfly arg-max.
// Phase 3 per row: softmax(top6), gather e_t, tanh gate, ka softmax, e_Nh;
//   emits (e_h+e_Nh) and (e_h*e_Nh) as f16 GEMM operands.
// ---------------------------------------------------------------------------
__device__ __forceinline__ void ins6(float* v, int* ix, float nv, int ni) {
  if (!(nv > v[5])) return;
  v[5] = nv; ix[5] = ni;
#pragma unroll
  for (int j = 5; j > 0; j--) {
    if (v[j] > v[j - 1]) {
      float tv = v[j]; v[j] = v[j - 1]; v[j - 1] = tv;
      int ti = ix[j]; ix[j] = ix[j - 1]; ix[j - 1] = ti;
    }
  }
}

__global__ __launch_bounds__(512) void attn_topk_agg(
    const _Float16* __restrict__ ehs16,  // [N,512] e_h*SCALE f16
    const _Float16* __restrict__ et16,   // [N,512] e_t f16
    const float* __restrict__ eH,        // [N,512] e_h f32
    const float* __restrict__ eT,        // [N,512] e_t f32
    _Float16* __restrict__ A16,          // out: e_h + e_Nh (f16)
    _Float16* __restrict__ M16) {        // out: e_h * e_Nh (f16)
  __shared__ _Float16 sA[16 * 520];      // 16 rows x 512 halfs, +8 half pad/row
  __shared__ float sL[16 * 516];         // logit chunk, +4 f32 pad/row
  __shared__ float sTopV[16][TOPK];
  __shared__ int   sTopI[16][TOPK];

  const int tid = threadIdx.x;
  const int wave = tid >> 5;
  const int lane = tid & 31;
  const int half = lane >> 4;
  const int l16  = lane & 15;
  const int rblk = blockIdx.x * 16;

  // stage the 16 A rows via the Tensor Data Mover (padding -> 520-half stride)
  if (wave == 0) {
    tdm_stage_rows(ehs16 + (size_t)rblk * DH, (void*)sA);
    __builtin_amdgcn_s_wait_tensorcnt(0);
  }
  if (tid < 16 * TOPK) {
    sTopV[tid / TOPK][tid % TOPK] = NEG_BIG;
    sTopI[tid / TOPK][tid % TOPK] = -1;
  }
  __syncthreads();

  for (int ch = 0; ch < N_NODES; ch += CHUNK) {
    // ---- compute 16 x CHUNK logits via WMMA: wave -> 2 adjacent tiles ----
    {
      const int tbase = wave * 32;  // column offset of this wave's 2 tiles
      const _Float16* bRow0 = et16 + (size_t)(ch + tbase + l16) * DH;
      const _Float16* bRow1 = et16 + (size_t)(ch + tbase + 16 + l16) * DH;
      const _Float16* aRow = sA + l16 * 520;
      v8f c0 = v8f{0.f}, c1 = v8f{0.f};
      v16h aP[2], b0P[2], b1P[2];
      aP[0]  = load_frag_a(aRow, 0, half);
      b0P[0] = load_frag_b(bRow0, 0, half);
      b1P[0] = load_frag_b(bRow1, 0, half);
      for (int k0 = 0; k0 < DH; k0 += 64) {
        const int k1 = k0 + 32;
        aP[1]  = load_frag_a(aRow, k1, half);
        b0P[1] = load_frag_b(bRow0, k1, half);
        b1P[1] = load_frag_b(bRow1, k1, half);
        c0 = __builtin_amdgcn_wmma_f32_16x16x32_f16(false, aP[0], false, b0P[0],
                                                    (short)0, c0, false, false);
        c1 = __builtin_amdgcn_wmma_f32_16x16x32_f16(false, aP[0], false, b1P[0],
                                                    (short)0, c1, false, false);
        const int k2 = (k0 + 64 < DH) ? k0 + 64 : 0;
        aP[0]  = load_frag_a(aRow, k2, half);
        b0P[0] = load_frag_b(bRow0, k2, half);
        b1P[0] = load_frag_b(bRow1, k2, half);
        c0 = __builtin_amdgcn_wmma_f32_16x16x32_f16(false, aP[1], false, b0P[1],
                                                    (short)0, c0, false, false);
        c1 = __builtin_amdgcn_wmma_f32_16x16x32_f16(false, aP[1], false, b1P[1],
                                                    (short)0, c1, false, false);
      }
#pragma unroll
      for (int v = 0; v < 8; v++) {
        sL[(v + 8 * half) * 516 + tbase + l16] = c0[v];
        sL[(v + 8 * half) * 516 + tbase + 16 + l16] = c1[v];
      }
    }
    __syncthreads();

    // ---- per-row top-6 merge: wave w owns row w ----
    {
      const int r = wave;
      float lv[TOPK]; int li[TOPK];
#pragma unroll
      for (int j = 0; j < TOPK; j++) { lv[j] = NEG_BIG; li[j] = -1; }
      if (lane == 0) {   // fold the running list in as candidates
#pragma unroll
        for (int j = 0; j < TOPK; j++) ins6(lv, li, sTopV[r][j], sTopI[r][j]);
      }
      for (int t2 = 0; t2 < CHUNK / 32; t2++) {
        const int cL = t2 * 32 + lane;
        ins6(lv, li, sL[r * 516 + cL], ch + cL);
      }
      // 6 arg-max extractions (wave32 butterfly)
      int ptr = 0;
      float fv[TOPK]; int fi[TOPK];
      for (int j = 0; j < TOPK; j++) {
        float bv = (ptr < TOPK) ? lv[ptr] : NEG_BIG;
        int bi = (ptr < TOPK) ? li[ptr] : -1;
#pragma unroll
        for (int off = 16; off >= 1; off >>= 1) {
          float ov = __shfl_xor(bv, off, 32);
          int oi = __shfl_xor(bi, off, 32);
          if (ov > bv || (ov == bv && oi != -1 && (unsigned)oi < (unsigned)bi)) {
            bv = ov; bi = oi;
          }
        }
        fv[j] = bv; fi[j] = bi;
        if (ptr < TOPK && li[ptr] == bi) ptr++;
      }
      if (lane == 0) {
#pragma unroll
        for (int j = 0; j < TOPK; j++) { sTopV[wave][j] = fv[j]; sTopI[wave][j] = fi[j]; }
      }
    }
    __syncthreads();
  }

  // ---- phase 3: wave w -> row rblk+w ----
  const int r = rblk + wave;
  float tv[TOPK]; int ti[TOPK];
#pragma unroll
  for (int j = 0; j < TOPK; j++) { tv[j] = sTopV[wave][j]; ti[j] = sTopI[wave][j]; }

  // softmax over top-k logits (list is sorted descending)
  float p[TOPK], ps = 0.f;
#pragma unroll
  for (int j = 0; j < TOPK; j++) { p[j] = __expf(tv[j] - tv[0]); ps += p[j]; }
  const float pinv = 1.f / ps;
#pragma unroll
  for (int j = 0; j < TOPK; j++) p[j] *= pinv;

  const float* ehRow = eH + (size_t)r * DH;
  float acc[TOPK] = {0.f, 0.f, 0.f, 0.f, 0.f, 0.f};
  for (int t = 0; t < DH / 32; t++) {
    const int d = lane + 32 * t;
    const float eh = ehRow[d];
#pragma unroll
    for (int j = 0; j < TOPK; j++) {
      const float nb = eT[(size_t)ti[j] * DH + d];
      const float ehr = p[j] * nb + (1.f - p[j]) * eh;
      const float g = tanhf(eh + ehr);
      acc[j] += nb * g;
    }
  }
#pragma unroll
  for (int j = 0; j < TOPK; j++) {
#pragma unroll
    for (int off = 16; off >= 1; off >>= 1) acc[j] += __shfl_xor(acc[j], off, 32);
  }
  float km = acc[0];
#pragma unroll
  for (int j = 1; j < TOPK; j++) km = fmaxf(km, acc[j]);
  float kp[TOPK], ks = 0.f;
#pragma unroll
  for (int j = 0; j < TOPK; j++) { kp[j] = __expf(acc[j] - km); ks += kp[j]; }
  const float kinv = 1.f / ks;
#pragma unroll
  for (int j = 0; j < TOPK; j++) kp[j] *= kinv;

  for (int t = 0; t < DH / 32; t++) {
    const int d = lane + 32 * t;
    const float eh = ehRow[d];
    float eN = 0.f;
#pragma unroll
    for (int j = 0; j < TOPK; j++) eN += kp[j] * eT[(size_t)ti[j] * DH + d];
    A16[(size_t)r * DH + d] = (_Float16)(eh + eN);
    M16[(size_t)r * DH + d] = (_Float16)(eh * eN);
  }
}

// ---------------------------------------------------------------------------
// readout: h = colsum(emb)/N -> LayerNorm -> h @ cls_w + cls_b. One block.
// ---------------------------------------------------------------------------
__global__ __launch_bounds__(512) void final_head(
    const float* __restrict__ colsumE, const float* __restrict__ ln_g,
    const float* __restrict__ ln_b, const float* __restrict__ cls_w,
    const float* __restrict__ cls_b, float* __restrict__ out) {
  __shared__ float red[DH];
  const int t = threadIdx.x;
  const float hv = colsumE[t] * (1.f / (float)N_NODES);
  red[t] = hv; __syncthreads();
  for (int s = 256; s > 0; s >>= 1) { if (t < s) red[t] += red[t + s]; __syncthreads(); }
  const float mu = red[0] * (1.f / (float)DH);
  __syncthreads();
  const float dv = hv - mu;
  red[t] = dv * dv; __syncthreads();
  for (int s = 256; s > 0; s >>= 1) { if (t < s) red[t] += red[t + s]; __syncthreads(); }
  const float var = red[0] * (1.f / (float)DH);
  __syncthreads();
  const float hn = dv * rsqrtf(var + 1e-5f) * ln_g[t] + ln_b[t];
  for (int c = 0; c < 2; c++) {
    red[t] = hn * cls_w[t * 2 + c]; __syncthreads();
    for (int s = 256; s > 0; s >>= 1) { if (t < s) red[t] += red[t + s]; __syncthreads(); }
    if (t == 0) out[c] = red[0] + cls_b[c];
    __syncthreads();
  }
}

// ---------------------------------------------------------------------------
extern "C" void kernel_launch(void* const* d_in, const int* in_sizes, int n_in,
                              void* d_out, int out_size, void* d_ws, size_t ws_size,
                              hipStream_t stream) {
  (void)in_sizes; (void)n_in; (void)out_size; (void)ws_size;
  const float* feats = (const float*)d_in[0];
  const float* fc1_w = (const float*)d_in[1];
  const float* fc1_b = (const float*)d_in[2];
  const float* wh_w  = (const float*)d_in[3];
  const float* wh_b  = (const float*)d_in[4];
  const float* wt_w  = (const float*)d_in[5];
  const float* wt_b  = (const float*)d_in[6];
  const float* l1_w  = (const float*)d_in[7];
  const float* l1_b  = (const float*)d_in[8];
  const float* l2_w  = (const float*)d_in[9];
  const float* l2_b  = (const float*)d_in[10];
  const float* ln_g  = (const float*)d_in[11];
  const float* ln_b  = (const float*)d_in[12];
  const float* cls_w = (const float*)d_in[13];
  const float* cls_b = (const float*)d_in[14];
  float* out = (float*)d_out;

  // workspace carve-up (256B aligned)
  char* ws = (char*)d_ws;
  size_t off = 0;
  auto carve = [&](size_t bytes) -> char* {
    char* p = ws + off;
    off += (bytes + 255) & ~(size_t)255;
    return p;
  };
  _Float16* feats16 = (_Float16*)carve((size_t)N_NODES * DIN * 2);
  _Float16* fc1wt   = (_Float16*)carve((size_t)DIN * DH * 2);
  _Float16* whwt    = (_Float16*)carve((size_t)DH * DH * 2);
  _Float16* wtwt    = (_Float16*)carve((size_t)DH * DH * 2);
  _Float16* l1wt    = (_Float16*)carve((size_t)DH * DH * 2);
  _Float16* l2wt    = (_Float16*)carve((size_t)DH * DH * 2);
  float*    x       = (float*)   carve((size_t)N_NODES * DH * 4);
  _Float16* x16     = (_Float16*)carve((size_t)N_NODES * DH * 2);
  float*    eh      = (float*)   carve((size_t)N_NODES * DH * 4);
  _Float16* ehs16   = (_Float16*)carve((size_t)N_NODES * DH * 2);
  float*    et      = (float*)   carve((size_t)N_NODES * DH * 4);
  _Float16* et16    = (_Float16*)carve((size_t)N_NODES * DH * 2);
  _Float16* a16     = (_Float16*)carve((size_t)N_NODES * DH * 2);
  _Float16* m16     = (_Float16*)carve((size_t)N_NODES * DH * 2);
  float*    emb     = (float*)   carve((size_t)N_NODES * DH * 4);
  float*    colsumX = (float*)   carve(DH * 4);
  float*    colsumE = (float*)   carve(DH * 4);

  // prep
  zero_f32<<<2, 512, 0, stream>>>(colsumX, DH);
  zero_f32<<<2, 512, 0, stream>>>(colsumE, DH);
  to_f16<<<(N_NODES * DIN + 255) / 256, 256, 0, stream>>>(feats, feats16, N_NODES * DIN);
  transpose_f16<<<(DIN * DH + 255) / 256, 256, 0, stream>>>(fc1_w, fc1wt, DIN, DH);
  transpose_f16<<<(DH * DH + 255) / 256, 256, 0, stream>>>(wh_w, whwt, DH, DH);
  transpose_f16<<<(DH * DH + 255) / 256, 256, 0, stream>>>(wt_w, wtwt, DH, DH);
  transpose_f16<<<(DH * DH + 255) / 256, 256, 0, stream>>>(l1_w, l1wt, DH, DH);
  transpose_f16<<<(DH * DH + 255) / 256, 256, 0, stream>>>(l2_w, l2wt, DH, DH);

  const dim3 gemmGrid(DH / 64, N_NODES / 128);
  // x = lrelu(feats @ fc1 + b)
  gemm_wmma<<<gemmGrid, 256, 0, stream>>>(feats16, fc1wt, fc1_b, x, nullptr, 1.f,
                                          N_NODES, DH, DIN, 1, 0);
  // x = (x + colmean(x)) * 0.5 ; also x16
  colsum_kernel<<<N_NODES / 64, 512, 0, stream>>>(x, colsumX);
  apply_mean<<<N_NODES, 512, 0, stream>>>(x, x16, colsumX);
  // e_h (f32) + e_h*SCALE (f16) ; e_t (f32 + f16)
  gemm_wmma<<<gemmGrid, 256, 0, stream>>>(x16, whwt, wh_b, eh, ehs16, ATT_SCALE,
                                          N_NODES, DH, DH, 0, 0);
  gemm_wmma<<<gemmGrid, 256, 0, stream>>>(x16, wtwt, wt_b, et, et16, 1.f,
                                          N_NODES, DH, DH, 0, 0);
  // fused attention / top-k / aggregation
  attn_topk_agg<<<N_NODES / 16, 512, 0, stream>>>(ehs16, et16, eh, et, a16, m16);
  // emb = lrelu(a@l1+b1) + lrelu(m@l2+b2)
  gemm_wmma<<<gemmGrid, 256, 0, stream>>>(a16, l1wt, l1_b, emb, nullptr, 1.f,
                                          N_NODES, DH, DH, 1, 0);
  gemm_wmma<<<gemmGrid, 256, 0, stream>>>(m16, l2wt, l2_b, emb, nullptr, 1.f,
                                          N_NODES, DH, DH, 1, 1);
  // readout
  colsum_kernel<<<N_NODES / 64, 512, 0, stream>>>(emb, colsumE);
  final_head<<<1, 512, 0, stream>>>(colsumE, ln_g, ln_b, cls_w, cls_b, out);
}
